// Encoder_214748365313
// MI455X (gfx1250) — matrix operand order repaired and checked by reading, not compiled, not run on hardware
//
#include <hip/hip_runtime.h>

typedef __attribute__((ext_vector_type(16))) _Float16 v16h;
typedef __attribute__((ext_vector_type(8)))  float    v8f;

#define NBATCH 32

static __device__ __forceinline__ void atomAddF(float* p, float v) {
  unsafeAtomicAdd(p, v);
}

// ---------------------------------------------------------------- utility
__global__ void fill_kernel(float* __restrict__ p, float v, long n) {
  long i = (long)blockIdx.x * blockDim.x + threadIdx.x;
  if (i < n) p[i] = v;
}

__global__ void bias_fill_kernel(float* __restrict__ out, const float* __restrict__ bias,
                                 int F, long n) {
  long i = (long)blockIdx.x * blockDim.x + threadIdx.x;
  if (i < n) out[i] = bias[(int)(i % F)];
}

__global__ void neg_copy_kernel(const float* __restrict__ x, float* __restrict__ y, long n) {
  long i = (long)blockIdx.x * blockDim.x + threadIdx.x;
  if (i < n) y[i] = -x[i];
}

// ------------------------------------------------------------ graph setup
__global__ void deg_kernel(const int* __restrict__ row, float* __restrict__ deg, int ne) {
  int i = blockIdx.x * blockDim.x + threadIdx.x;
  if (i < ne) atomAddF(&deg[row[i]], 1.0f);
}

__global__ void dinv_kernel(float* __restrict__ deg, int n) {
  int i = blockIdx.x * blockDim.x + threadIdx.x;
  if (i < n) {
    float d = deg[i];
    deg[i] = (d > 0.0f) ? rsqrtf(d) : 0.0f;
  }
}

__global__ void edgew_kernel(const int* __restrict__ row, const int* __restrict__ col,
                             const float* __restrict__ dinv, float* __restrict__ w, int ne) {
  int i = blockIdx.x * blockDim.x + threadIdx.x;
  if (i < ne) w[i] = -dinv[row[i]] * dinv[col[i]];
}

// scatter SpMM:  y[b, rows[e], f] += beta * w[e] * x[b, cols[e], f]
__global__ void spmm_kernel(const int* __restrict__ rows, const int* __restrict__ cols,
                            const float* __restrict__ w, float beta,
                            const float* __restrict__ x, float* __restrict__ y,
                            int F, int Nin, int Nout, long total) {
  long i = (long)blockIdx.x * blockDim.x + threadIdx.x;
  if (i >= total) return;
  int BF = NBATCH * F;
  int e = (int)(i / BF);
  int j = (int)(i % BF);
  int b = j / F, f = j - b * F;
  float val = beta * w[e] * x[((long)b * Nin + cols[e]) * F + f];
  atomAddF(&y[((long)b * Nout + rows[e]) * F + f], val);
}

// -------------------------------------------- fragment packing (f16 WMMA)
// A fragment layout: A16[((tile*ksteps_total + ks)*32 + lane)*16 + i]
// A-lane map (16x32 f16): K = ks*32 + i + 8g (+8 when i>=8)
__global__ void pack_a_full(const float* __restrict__ A, _Float16* __restrict__ A16,
                            int Fin, int ksteps_total, int ks_base, int ksteps_k, long total) {
  long idx = (long)blockIdx.x * blockDim.x + threadIdx.x;
  if (idx >= total) return;
  int  i    = (int)(idx & 15);
  int  lane = (int)((idx >> 4) & 31);
  long t    = idx >> 9;
  int  ks   = (int)(t % ksteps_k);
  long tile = t / ksteps_k;
  int g = lane >> 4, lm = lane & 15;
  long m = tile * 16 + lm;
  int ka = ks * 32 + ((i < 8) ? (i + 8 * g) : (i + 8 + 8 * g));
  A16[((tile * ksteps_total + ks_base + ks) * 32 + lane) * 16 + i] = (_Float16)A[m * (long)Fin + ka];
}

// level-0 partial pack: scatter [M,3] into K offsets kaoff..kaoff+2 of k-step 0
__global__ void pack_a_l0(const float* __restrict__ A, _Float16* __restrict__ A16,
                          int kaoff, long total) {
  long idx = (long)blockIdx.x * blockDim.x + threadIdx.x;
  if (idx >= total) return;
  int  j    = (int)(idx % 3);
  long r    = idx / 3;
  int  lm   = (int)(r & 15);
  long tile = r >> 4;
  int ka = kaoff + j;
  int g = (ka >> 3) & 1;
  int i = (ka & 7) | ((ka >> 4) << 3);
  int lane = g * 16 + lm;
  A16[(tile * 32 + lane) * 16 + i] = (_Float16)A[(tile * 16 + lm) * 3 + j];
}

// B fragment layout: B16[((ks*ntiles + nt)*32 + lane)*16 + i]
// B-lane map (32x16 f16): K = ks*32 + i + 8g (+8 when i>=8); W viewed as [Ktrue, Fout] flat
__global__ void pack_b_kernel(const float* __restrict__ W, _Float16* __restrict__ B16,
                              long Ktrue, int Fout, int ntiles, long total) {
  long idx = (long)blockIdx.x * blockDim.x + threadIdx.x;
  if (idx >= total) return;
  int  i    = (int)(idx & 15);
  int  lane = (int)((idx >> 4) & 31);
  long t    = idx >> 9;
  int  nt   = (int)(t % ntiles);
  long ks   = t / ntiles;
  int g = lane >> 4, lm = lane & 15;
  long kg = ks * 32 + ((i < 8) ? (i + 8 * g) : (i + 8 + 8 * g));
  long kc = (kg < Ktrue) ? kg : 0;
  float v = W[kc * Fout + nt * 16 + lm];
  B16[idx] = (_Float16)((kg < Ktrue) ? v : 0.0f);
}

// ------------------------------------- WMMA GEMM: OUT = A16 @ B16 + bias
__global__ void gemm_wmma_bias(const _Float16* __restrict__ A16, const _Float16* __restrict__ B16,
                               const float* __restrict__ bias, float* __restrict__ OUT,
                               long mtiles, int ntiles, int ksteps, int Fout) {
  long wave = ((long)blockIdx.x * blockDim.x + threadIdx.x) >> 5;
  int lane = threadIdx.x & 31;
  long m_tile = wave / ntiles;
  int  n_tile = (int)(wave - m_tile * ntiles);
  if (m_tile >= mtiles) return;              // wave-uniform
  int g = lane >> 4, lm = lane & 15;
  const v16h* Af = (const v16h*)A16 + m_tile * ksteps * 32 + lane;
  const v16h* Bf = (const v16h*)B16 + (long)n_tile * 32 + lane;
  v8f c = {};
  for (int ks = 0; ks < ksteps; ++ks) {
    v16h a = Af[(long)ks * 32];
    v16h b = Bf[(long)ks * ntiles * 32];
    c = __builtin_amdgcn_wmma_f32_16x16x32_f16(false, a, false, b, (short)0, c, false, false);
  }
  float bs = bias[n_tile * 16 + lm];
  float* Orow = OUT + m_tile * 16 * (long)Fout + n_tile * 16 + lm;
#pragma unroll
  for (int r = 0; r < 8; ++r) Orow[(long)(r + 8 * g) * Fout] = c[r] + bs;
}

// ----------------------------------------------------- BatchNorm over nodes
__global__ void bn_stats_kernel(const float* __restrict__ out, float* __restrict__ mean,
                                float* __restrict__ rstd, int N, int F) {
  int n = blockIdx.x;
  int BF = NBATCH * F;
  float s = 0.0f, s2 = 0.0f;
  for (int j = threadIdx.x; j < BF; j += blockDim.x) {
    int b = j / F, f = j - b * F;
    float v = out[((long)b * N + n) * F + f];
    s += v; s2 += v * v;
  }
  __shared__ float sh[256], sh2[256];
  sh[threadIdx.x] = s; sh2[threadIdx.x] = s2;
  __syncthreads();
  for (int st = 128; st > 0; st >>= 1) {
    if (threadIdx.x < st) { sh[threadIdx.x] += sh[threadIdx.x + st]; sh2[threadIdx.x] += sh2[threadIdx.x + st]; }
    __syncthreads();
  }
  if (threadIdx.x == 0) {
    float m = sh[0] / (float)BF;
    float v = sh2[0] / (float)BF - m * m;
    mean[n] = m;
    rstd[n] = rsqrtf(v + 1e-5f);
  }
}

__global__ void bn_elu_kernel(float* __restrict__ out, const float* __restrict__ mean,
                              const float* __restrict__ rstd, const float* __restrict__ gamma,
                              const float* __restrict__ beta, int N, int F, long total) {
  long i = (long)blockIdx.x * blockDim.x + threadIdx.x;
  if (i >= total) return;
  int n = (int)((i / F) % N);
  float v = (out[i] - mean[n]) * rstd[n] * gamma[n] + beta[n];
  out[i] = (v > 0.0f) ? v : (expf(v) - 1.0f);
}

// --------------------------------------------- final FC, split-K WMMA
__global__ void fc_wmma_kernel(const _Float16* __restrict__ A16, const _Float16* __restrict__ B16,
                               float* __restrict__ out, int ksteps_total, int kchunk) {
  int wave = (int)(((long)blockIdx.x * blockDim.x + threadIdx.x) >> 5);
  int lane = threadIdx.x & 31;
  int m_tile = wave & 1;
  int n_tile = (wave >> 1) & 7;
  int kc = wave >> 4;
  int ks0 = kc * kchunk;
  if (ks0 >= ksteps_total) return;
  int ks1 = ks0 + kchunk; if (ks1 > ksteps_total) ks1 = ksteps_total;
  int g = lane >> 4, lm = lane & 15;
  const v16h* Af = (const v16h*)A16 + (long)m_tile * ksteps_total * 32 + lane;
  const v16h* Bf = (const v16h*)B16 + (long)n_tile * 32 + lane;
  v8f c = {};
  for (int ks = ks0; ks < ks1; ++ks) {
    v16h a = Af[(long)ks * 32];
    v16h b = Bf[(long)ks * 8 * 32];
    c = __builtin_amdgcn_wmma_f32_16x16x32_f16(false, a, false, b, (short)0, c, false, false);
  }
#pragma unroll
  for (int r = 0; r < 8; ++r)
    atomAddF(&out[(m_tile * 16 + r + 8 * g) * 128 + n_tile * 16 + lm], c[r]);
}

// ================================================================== host
static inline long cdivl(long a, long b) { return (a + b - 1) / b; }

extern "C" void kernel_launch(void* const* d_in, const int* in_sizes, int n_in,
                              void* d_out, int out_size, void* d_ws, size_t ws_size,
                              hipStream_t stream) {
  (void)in_sizes; (void)n_in; (void)out_size; (void)ws_size;
  const int NL[4]   = {35709, 8928, 2232, 558};
  const int FIN[3]  = {3, 32, 64};
  const int FOUT[3] = {32, 64, 128};

  const float* x0     = (const float*)d_in[0];
  const int*   ei[3]  = {(const int*)d_in[1], (const int*)d_in[2], (const int*)d_in[3]};
  const int*   prow[3]= {(const int*)d_in[4], (const int*)d_in[7], (const int*)d_in[10]};
  const int*   pcol[3]= {(const int*)d_in[5], (const int*)d_in[8], (const int*)d_in[11]};
  const float* pval[3]= {(const float*)d_in[6], (const float*)d_in[9], (const float*)d_in[12]};
  const float* W[3]   = {(const float*)d_in[13], (const float*)d_in[17], (const float*)d_in[21]};
  const float* bb[3]  = {(const float*)d_in[14], (const float*)d_in[18], (const float*)d_in[22]};
  const float* gam[3] = {(const float*)d_in[15], (const float*)d_in[19], (const float*)d_in[23]};
  const float* bet[3] = {(const float*)d_in[16], (const float*)d_in[20], (const float*)d_in[24]};
  const float* fcw    = (const float*)d_in[25];
  const float* fcb    = (const float*)d_in[26];
  float* out = (float*)d_out;

  // --------- scratch arena (units: f32), all offsets 64-f32 (256B) aligned
  float* base = (float*)d_ws;
  long o = 0;
  auto take = [&](long nf32) { float* p = base + o; o += (nf32 + 63) & ~63L; return p; };
  const long CAP_TX = 32L * 8928 * 32;                 // 9,142,272
  float* TXB[3] = { take(CAP_TX), take(CAP_TX), take(CAP_TX) };
  float* OUT  = take(32L * 35709 * 32);                // 36,566,016
  float* X1   = take(32L * 8928 * 32);
  float* X2   = take(32L * 2232 * 64);
  float* X3   = take(32L * 558 * 128);
  float* EW   = take(6L * 35709);
  float* DEG  = take(35709);
  float* MEAN = take(35709);
  float* RSTD = take(35709);
  _Float16* A16   = (_Float16*)take(27426816);         // max: lvl1 = 17856*6*512 halves
  _Float16* B16   = (_Float16*)take(12 * 8 * 512 / 2);
  _Float16* FCA16 = (_Float16*)take(2232L * 2 * 512 / 2);
  _Float16* FCB16 = (_Float16*)take(2232L * 8 * 512 / 2);
  float* XNEXT[3] = { X1, X2, X3 };

  const int T = 256;
  const float* xl = x0;

  for (int lvl = 0; lvl < 3; ++lvl) {
    const int N = NL[lvl], Nn = NL[lvl + 1], fi = FIN[lvl], fo = FOUT[lvl];
    const int ne = 6 * N, nnz = 3 * Nn;
    const int* row = ei[lvl];
    const int* col = ei[lvl] + ne;
    const long M = 32L * N;
    const long mtiles = M >> 4;
    const int  ntiles = fo >> 4;
    const long outsz = M * fo;
    const long txsz  = M * fi;
    const int  ksteps_k     = (lvl == 0) ? 1 : (fi >> 5);   // k-steps per cheb term
    const int  ksteps_total = (lvl == 0) ? 1 : 6 * ksteps_k;
    const long Ktrue        = (lvl == 0) ? 18 : (long)6 * fi;

    // ---- edge weights: -(d^-1/2 A d^-1/2)
    fill_kernel<<<(int)cdivl(N, T), T, 0, stream>>>(DEG, 0.0f, N);
    deg_kernel<<<(int)cdivl(ne, T), T, 0, stream>>>(row, DEG, ne);
    dinv_kernel<<<(int)cdivl(N, T), T, 0, stream>>>(DEG, N);
    edgew_kernel<<<(int)cdivl(ne, T), T, 0, stream>>>(row, col, DEG, EW, ne);

    // ---- pack weights into B fragments (W flat is [6*fi, fo]; lvl0 true K=18)
    long btotal = (long)ksteps_total * ntiles * 512;
    pack_b_kernel<<<(int)cdivl(btotal, T), T, 0, stream>>>(W[lvl], B16, Ktrue, fo, ntiles, btotal);

    // ---- A fragment buffer; level 0 needs zero-fill for K padding 18->32
    long ahalves = mtiles * ksteps_total * 512;
    if (lvl == 0)
      fill_kernel<<<(int)cdivl(ahalves / 2, T), T, 0, stream>>>((float*)A16, 0.0f, ahalves / 2);

    auto packA = [&](const float* src, int k) {
      if (lvl == 0) {
        long tot = mtiles * 48;
        pack_a_l0<<<(int)cdivl(tot, T), T, 0, stream>>>(src, A16, 3 * k, tot);
      } else {
        long tot = mtiles * ksteps_k * 512;
        pack_a_full<<<(int)cdivl(tot, T), T, 0, stream>>>(src, A16, fi, ksteps_total,
                                                          k * ksteps_k, ksteps_k, tot);
      }
    };

    // ---- Chebyshev recurrence, packing each basis as it is produced
    packA(xl, 0);                                                // T0 = x
    fill_kernel<<<(int)cdivl(txsz, T), T, 0, stream>>>(TXB[1], 0.0f, txsz);
    {
      long tot = (long)ne * 32 * fi;                             // T1 = Lhat(T0)
      spmm_kernel<<<(int)cdivl(tot, T), T, 0, stream>>>(row, col, EW, 1.0f, xl, TXB[1], fi, N, N, tot);
    }
    packA(TXB[1], 1);
    const float* t0 = xl;
    const float* t1 = TXB[1];
    float* targets[4] = { TXB[2], TXB[0], TXB[1], TXB[2] };
    for (int k = 2; k < 6; ++k) {                                // T2 = 2*Lhat(T1) - T0
      float* t2 = targets[k - 2];
      neg_copy_kernel<<<(int)cdivl(txsz, T), T, 0, stream>>>(t0, t2, txsz);
      long tot = (long)ne * 32 * fi;
      spmm_kernel<<<(int)cdivl(tot, T), T, 0, stream>>>(row, col, EW, 2.0f, t1, t2, fi, N, N, tot);
      packA(t2, k);
      t0 = t1; t1 = t2;
    }

    // ---- single fused GEMM over concatenated basis: OUT = A @ W + b
    long gemm_waves = mtiles * ntiles;
    gemm_wmma_bias<<<(int)cdivl(gemm_waves * 32, T), T, 0, stream>>>(
        A16, B16, bb[lvl], OUT, mtiles, ntiles, ksteps_total, fo);

    // ---- BatchNorm over nodes + ELU (in place)
    bn_stats_kernel<<<N, 256, 0, stream>>>(OUT, MEAN, RSTD, N, fo);
    bn_elu_kernel<<<(int)cdivl(outsz, T), T, 0, stream>>>(OUT, MEAN, RSTD, gam[lvl], bet[lvl], N, fo, outsz);

    // ---- pool to next level
    float* xn = XNEXT[lvl];
    long xnsz = 32L * Nn * fo;
    fill_kernel<<<(int)cdivl(xnsz, T), T, 0, stream>>>(xn, 0.0f, xnsz);
    long ptot = (long)nnz * 32 * fo;
    spmm_kernel<<<(int)cdivl(ptot, T), T, 0, stream>>>(prow[lvl], pcol[lvl], pval[lvl], 1.0f, OUT, xn, fo, N, Nn, ptot);
    xl = xn;
  }

  // ---- final FC: [32, 71424] @ [71424, 128] + fc_b   (K-steps = 2232)
  {
    const int kst = 2232;
    long atot = 2L * kst * 512;
    pack_a_full<<<(int)cdivl(atot, T), T, 0, stream>>>(X3, FCA16, 71424, kst, 0, kst, atot);
    long btot = (long)kst * 8 * 512;
    pack_b_kernel<<<(int)cdivl(btot, T), T, 0, stream>>>(fcw, FCB16, 71424, 128, 8, btot);
    bias_fill_kernel<<<16, T, 0, stream>>>(out, fcb, 128, 32L * 128);
    const int kchunk = 18;                       // 124 chunks exactly
    const int fc_waves = 16 * (kst / kchunk);
    fc_wmma_kernel<<<(int)cdivl((long)fc_waves * 32, T), T, 0, stream>>>(FCA16, FCB16, out, kst, kchunk);
  }
}
